// Transfusion_56556129353904
// MI455X (gfx1250) — matrix operand order repaired and checked by reading, not compiled, not run on hardware
//
#include <hip/hip_runtime.h>
#include <math.h>

#define HEADS 16
#define DHEAD 64
#define NSEQ  2048
#define BATCH 2
#define DMODEL 1024
#define NMOD  4
#define NEG_BIG (-1.0e30f)

typedef __bf16 bf16;
typedef __attribute__((ext_vector_type(16))) __bf16 v16bf;
typedef __attribute__((ext_vector_type(8)))  float  v8f;

union Frag { unsigned u[8]; v16bf v; };

__device__ __forceinline__ v8f wmma_bf16(v16bf a, v16bf b, v8f c) {
  return __builtin_amdgcn_wmma_f32_16x16x32_bf16(false, a, false, b, (short)0, c, false, false);
}
// A-matrix (16x32 bf16) per-lane element->K map (contiguous: k = 16*hi + element, matches B fills)
__device__ __forceinline__ int a_kidx(int j, int hi) {
  return hi * 16 + 2 * j;
}

// gfx1250 async copy: memory -> LDS, 16B per lane, tracked by ASYNCcnt.
// Low 32 bits of a generic LDS pointer are the LDS byte offset (ISA 10.2).
__device__ __forceinline__ void async_load_b128(void* lds, const void* gptr) {
  asm volatile("global_load_async_to_lds_b128 %0, %1, off"
               :: "v"((unsigned)(uintptr_t)lds),
                  "v"((unsigned long long)(uintptr_t)gptr)
               : "memory");
}
__device__ __forceinline__ void wait_async0() {
  asm volatile("s_wait_asynccnt 0x0" ::: "memory");
}

// ---------------- f32 -> bf16 convert + transpose: in[K][N] -> out[N][K] ----------------
__global__ void convert_bf16_T(const float* __restrict__ in, bf16* __restrict__ out,
                               int K, int N) {
  int idx = blockIdx.x * 256 + threadIdx.x;
  if (idx >= K * N) return;
  int k = idx / N, n = idx - k * N;
  out[(size_t)n * K + k] = (bf16)in[idx];
}

// ---------------- prep: modality substitute + RMS norm + pos ----------------
__global__ __launch_bounds__(256)
void prep_kernel(const float* __restrict__ x, const float* __restrict__ tokens,
                 const float* __restrict__ gamma, const int* __restrict__ modal,
                 bf16* __restrict__ xn, int* __restrict__ pos) {
  int row = blockIdx.x;            // 0..4095
  int b = row >> 11, n = row & 2047;
  int tid = threadIdx.x;
  int off[NMOD], len[NMOD];
#pragma unroll
  for (int m = 0; m < NMOD; ++m) {
    off[m] = modal[(b * NMOD + m) * 3 + 1];
    len[m] = modal[(b * NMOD + m) * 3 + 2];
  }
  bool any = false;
#pragma unroll
  for (int m = 0; m < NMOD; ++m) any = any || (n >= off[m] && n < off[m] + len[m]);

  float vals[4];
  float ss = 0.f;
#pragma unroll
  for (int e = 0; e < 4; ++e) {
    int d = e * 256 + tid;
    float v;
    if (any) {
      v = 0.f;
#pragma unroll
      for (int m = 0; m < NMOD; ++m) {
        if (n >= off[m] && n < off[m] + len[m]) {
          int rel = n - off[m];
          v += tokens[(((size_t)(b * NMOD + m)) * 256 + rel) * DMODEL + d];
        }
      }
    } else {
      v = x[(size_t)row * DMODEL + d];
    }
    vals[e] = v;
    ss += v * v;
  }
  __shared__ float red[256];
  red[tid] = ss;
  __syncthreads();
  for (int s2 = 128; s2 > 0; s2 >>= 1) {
    if (tid < s2) red[tid] += red[tid + s2];
    __syncthreads();
  }
  float norm = sqrtf(red[0]);
  float sc = sqrtf((float)DMODEL) / fmaxf(norm, 1e-12f);
#pragma unroll
  for (int e = 0; e < 4; ++e) {
    int d = e * 256 + tid;
    xn[(size_t)row * DMODEL + d] = (bf16)(vals[e] * sc * (gamma[d] + 1.f));
  }
  if (tid == 0) {
    // pos = n - |union of [off+1, off+len) intersected with [0, n]|
    int s[NMOD], e[NMOD];
#pragma unroll
    for (int m = 0; m < NMOD; ++m) {
      s[m] = off[m] + 1;
      e[m] = off[m] + len[m];
      if (e[m] < s[m]) e[m] = s[m];
    }
    for (int a = 0; a < NMOD - 1; ++a)
      for (int m = 0; m < NMOD - 1 - a; ++m)
        if (s[m] > s[m + 1]) {
          int t0 = s[m]; s[m] = s[m + 1]; s[m + 1] = t0;
          int t1 = e[m]; e[m] = e[m + 1]; e[m + 1] = t1;
        }
    int total = 0, cs = s[0], ce = e[0];
#pragma unroll
    for (int m = 1; m < NMOD; ++m) {
      if (s[m] > ce) {
        int ov = min(ce, n + 1) - cs;
        if (ov > 0) total += ov;
        cs = s[m]; ce = e[m];
      } else if (e[m] > ce) ce = e[m];
    }
    int ov = min(ce, n + 1) - cs;
    if (ov > 0) total += ov;
    pos[row] = n - total;
  }
}

// ---------------- generic bf16 GEMM: C[MxN] = A[MxK] * BT[N][K]^T ----------------
// BT is pre-transposed; both LDS stages are async b128 direct-to-LDS copies.
template <bool OUT_BF16>
__global__ __launch_bounds__(256)
void gemm_bf16_kernel(const bf16* __restrict__ A, const bf16* __restrict__ BT,
                      void* __restrict__ Cout, int M, int N, int K) {
  __shared__ __align__(16) bf16 sA[128 * 32];    // [m][k]
  __shared__ __align__(16) bf16 sBT[128 * 32];   // [n][k]
  int tid = threadIdx.x;
  int lane = tid & 31, w = tid >> 5;
  int hi = lane >> 4, l15 = lane & 15;
  int wr = w >> 1, wc = w & 1;       // wave tile: rows wr*32, cols wc*64
  int m0 = blockIdx.y * 128, n0 = blockIdx.x * 128;

  v8f acc[2][4];
#pragma unroll
  for (int i = 0; i < 2; ++i)
#pragma unroll
    for (int j = 0; j < 4; ++j)
      acc[i][j] = (v8f){0.f, 0.f, 0.f, 0.f, 0.f, 0.f, 0.f, 0.f};

  for (int k0 = 0; k0 < K; k0 += 32) {
    __syncthreads();
    // stage A tile 128x32 and BT tile 128x32: async direct-to-LDS (ASYNCcnt)
#pragma unroll
    for (int it = 0; it < 2; ++it) {
      int i = tid + it * 256;
      int r = i >> 2, c8 = (i & 3) * 8;
      const bf16* ga = A  + (size_t)(m0 + r) * K + k0 + c8;
      const bf16* gb = BT + (size_t)(n0 + r) * K + k0 + c8;
      async_load_b128(&sA[r * 32 + c8], ga);
      async_load_b128(&sBT[r * 32 + c8], gb);
      if (k0 + 32 < K) {   // prefetch next k-step (global_prefetch_b8)
        __builtin_prefetch(ga + 32, 0, 1);
        __builtin_prefetch(gb + 32, 0, 1);
      }
    }
    wait_async0();
    __syncthreads();

    Frag af[2];
#pragma unroll
    for (int sm = 0; sm < 2; ++sm) {
      const bf16* ap = &sA[(wr * 32 + sm * 16 + l15) * 32];
#pragma unroll
      for (int j = 0; j < 8; ++j) af[sm].u[j] = *(const unsigned*)&ap[a_kidx(j, hi)];
    }
    Frag bfr[4];
#pragma unroll
    for (int sn = 0; sn < 4; ++sn) {
      const bf16* bp = &sBT[(wc * 64 + sn * 16 + l15) * 32];
#pragma unroll
      for (int j = 0; j < 8; ++j) bfr[sn].u[j] = *(const unsigned*)&bp[hi * 16 + 2 * j];
    }
#pragma unroll
    for (int sm = 0; sm < 2; ++sm)
#pragma unroll
      for (int sn = 0; sn < 4; ++sn)
        acc[sm][sn] = wmma_bf16(af[sm].v, bfr[sn].v, acc[sm][sn]);
  }

  float* Cf = (float*)Cout;
  bf16*  Cb = (bf16*)Cout;
#pragma unroll
  for (int sm = 0; sm < 2; ++sm)
#pragma unroll
    for (int sn = 0; sn < 4; ++sn) {
      size_t base = (size_t)(m0 + wr * 32 + sm * 16 + 8 * hi) * N +
                    (n0 + wc * 64 + sn * 16 + l15);
#pragma unroll
      for (int r = 0; r < 8; ++r) {
        float v = acc[sm][sn][r];
        if constexpr (OUT_BF16) Cb[base] = (bf16)v;
        else                    Cf[base] = v;
        base += (size_t)N;
      }
    }
}

// ---------------- RoPE + split into Q/K (bh, n, 64), V transposed (bh, 64, n) ----------------
__global__ __launch_bounds__(256)
void rope_split(const bf16* __restrict__ qkv, const int* __restrict__ pos,
                bf16* __restrict__ Q, bf16* __restrict__ K, bf16* __restrict__ VT) {
  int idx = blockIdx.x * 256 + threadIdx.x;       // b*n*h*32
  if (idx >= BATCH * NSEQ * HEADS * 32) return;
  int i = idx & 31;
  int t = idx >> 5;
  int h = t & 15; t >>= 4;
  int n = t & 2047;
  int b = t >> 11;
  size_t row = (size_t)(b * NSEQ + n) * 3072;
  int col = h * DHEAD + 2 * i;
  float q1 = (float)qkv[row + col],        q2 = (float)qkv[row + col + 1];
  float k1 = (float)qkv[row + 1024 + col], k2 = (float)qkv[row + 1024 + col + 1];
  float v1 = (float)qkv[row + 2048 + col], v2 = (float)qkv[row + 2048 + col + 1];
  float p = (float)pos[b * NSEQ + n];
  // freq = 10000^(-i/32)
  float freq = __expf(-(float)i * (9.210340371976184f / 32.f));
  float ang = p * freq, sn, cs;
  __sincosf(ang, &sn, &cs);
  int bh = b * HEADS + h;
  size_t orow = ((size_t)bh * NSEQ + n) * DHEAD + 2 * i;
  Q[orow]     = (bf16)(q1 * cs - q2 * sn);
  Q[orow + 1] = (bf16)(q1 * sn + q2 * cs);
  K[orow]     = (bf16)(k1 * cs - k2 * sn);
  K[orow + 1] = (bf16)(k1 * sn + k2 * cs);
  size_t vrow = ((size_t)bh * DHEAD + 2 * i) * NSEQ + n;   // [bh][dim][key]
  VT[vrow]        = (bf16)v1;
  VT[vrow + NSEQ] = (bf16)v2;
}

// ---------------- flash attention with softcap + modality mask ----------------
__global__ __launch_bounds__(128)
void attn_kernel(const bf16* __restrict__ Q, const bf16* __restrict__ K,
                 const bf16* __restrict__ VT, const int* __restrict__ modal,
                 bf16* __restrict__ O) {
  __shared__ __align__(16) bf16 sK[64 * 64];       // [key][dim]
  __shared__ __align__(16) bf16 sVT[64 * 64];      // [dim][key]
  __shared__ __align__(16) bf16 sP[4][16 * 64];    // per-wave P
  int tid = threadIdx.x, lane = tid & 31, wid = tid >> 5;
  int hi = lane >> 4, l15 = lane & 15;
  int bh = blockIdx.y;
  int b = bh >> 4, h = bh & 15;
  int q0 = blockIdx.x * 64 + wid * 16;
  const bf16* qp  = Q  + (size_t)bh * NSEQ * DHEAD;
  const bf16* kp  = K  + (size_t)bh * NSEQ * DHEAD;
  const bf16* vtp = VT + (size_t)bh * DHEAD * NSEQ;

  int off[NMOD], len[NMOD];
#pragma unroll
  for (int m = 0; m < NMOD; ++m) {
    off[m] = modal[(b * NMOD + m) * 3 + 1];
    len[m] = modal[(b * NMOD + m) * 3 + 2];
  }

  // Q fragments (two K-dim chunks of 32)
  Frag aq[2];
#pragma unroll
  for (int c = 0; c < 2; ++c)
#pragma unroll
    for (int j = 0; j < 8; ++j)
      aq[c].u[j] = *(const unsigned*)(qp + (size_t)(q0 + l15) * DHEAD + 32 * c + a_kidx(j, hi));

  float mrun[8], lrun[8];
  v8f o[4];
#pragma unroll
  for (int r = 0; r < 8; ++r) { mrun[r] = NEG_BIG; lrun[r] = 0.f; }
#pragma unroll
  for (int t = 0; t < 4; ++t) o[t] = (v8f){0.f, 0.f, 0.f, 0.f, 0.f, 0.f, 0.f, 0.f};

  const float scale = 0.125f;  // 64^-0.5
  for (int kt = 0; kt < NSEQ / 64; ++kt) {
    int kbase = kt * 64;
    __syncthreads();
    // stage K [key][dim] and VT [dim][key]: async direct-to-LDS (ASYNCcnt)
#pragma unroll
    for (int it = 0; it < 4; ++it) {
      int i = tid + it * 128;
      int r = i >> 3, c8 = (i & 7) * 8;
      const bf16* gk = kp  + (size_t)(kbase + r) * DHEAD + c8;
      const bf16* gv = vtp + (size_t)r * NSEQ + kbase + c8;
      async_load_b128(&sK[r * 64 + c8],  gk);
      async_load_b128(&sVT[r * 64 + c8], gv);
      if (kbase + 64 < NSEQ) {   // prefetch next key tile
        __builtin_prefetch(gk + (size_t)64 * DHEAD, 0, 1);
        __builtin_prefetch(gv + 64, 0, 1);
      }
    }
    wait_async0();
    __syncthreads();

    // scores: 4 sub-tiles of 16 keys
    v8f s[4];
#pragma unroll
    for (int kk = 0; kk < 4; ++kk) {
      v8f a = (v8f){0.f, 0.f, 0.f, 0.f, 0.f, 0.f, 0.f, 0.f};
#pragma unroll
      for (int c = 0; c < 2; ++c) {
        Frag bk;
        int key = kk * 16 + l15;
#pragma unroll
        for (int j = 0; j < 8; ++j)
          bk.u[j] = *(const unsigned*)&sK[key * 64 + 32 * c + hi * 16 + 2 * j];
        a = wmma_bf16(aq[c].v, bk.v, a);
      }
      s[kk] = a;
    }

    // softcap + mask + online softmax per row
#pragma unroll
    for (int r = 0; r < 8; ++r) {
      int i_glob = q0 + r + 8 * hi;
      float rowm = NEG_BIG;
#pragma unroll
      for (int kk = 0; kk < 4; ++kk) {
        int j_glob = kbase + kk * 16 + l15;
        float sv = s[kk][r] * scale;
        sv = 50.f * tanhf(sv * 0.02f);
        bool ok = (i_glob >= j_glob);
#pragma unroll
        for (int m = 0; m < NMOD; ++m)
          ok = ok || ((i_glob >= off[m]) && (j_glob < off[m] + len[m]));
        sv = ok ? sv : NEG_BIG;
        s[kk][r] = sv;
        rowm = fmaxf(rowm, sv);
      }
#pragma unroll
      for (int d = 8; d; d >>= 1) rowm = fmaxf(rowm, __shfl_xor(rowm, d, 16));
      float nm = fmaxf(mrun[r], rowm);
      float corr = __expf(mrun[r] - nm);
      mrun[r] = nm;
      float rsum = 0.f;
#pragma unroll
      for (int kk = 0; kk < 4; ++kk) {
        float p = __expf(s[kk][r] - nm);
        s[kk][r] = p;
        rsum += p;
      }
#pragma unroll
      for (int d = 8; d; d >>= 1) rsum += __shfl_xor(rsum, d, 16);
      lrun[r] = lrun[r] * corr + rsum;
#pragma unroll
      for (int t = 0; t < 4; ++t) o[t][r] *= corr;
#pragma unroll
      for (int kk = 0; kk < 4; ++kk)
        sP[wid][(r + 8 * hi) * 64 + kk * 16 + l15] = (bf16)s[kk][r];
    }
    __syncthreads();

    // O += P @ V
#pragma unroll
    for (int c = 0; c < 2; ++c) {
      Frag ap;
#pragma unroll
      for (int j = 0; j < 8; ++j)
        ap.u[j] = *(const unsigned*)&sP[wid][l15 * 64 + 32 * c + a_kidx(j, hi)];
#pragma unroll
      for (int t = 0; t < 4; ++t) {
        Frag bv;
#pragma unroll
        for (int j = 0; j < 8; ++j) {
          int key = 32 * c + hi * 16 + 2 * j;
          bv.u[j] = *(const unsigned*)&sVT[(t * 16 + l15) * 64 + key];
        }
        o[t] = wmma_bf16(ap.v, bv.v, o[t]);
      }
    }
  }

  // normalize and write (b, n, h*64) bf16
#pragma unroll
  for (int t = 0; t < 4; ++t) {
    size_t base = (size_t)(b * NSEQ + q0 + 8 * hi) * (HEADS * DHEAD) +
                  h * DHEAD + t * 16 + l15;
#pragma unroll
    for (int r = 0; r < 8; ++r) {
      O[base] = (bf16)(o[t][r] / lrun[r]);
      base += HEADS * DHEAD;
    }
  }
}

// ---------------- launch ----------------
extern "C" void kernel_launch(void* const* d_in, const int* in_sizes, int n_in,
                              void* d_out, int out_size, void* d_ws, size_t ws_size,
                              hipStream_t stream) {
  const float* x      = (const float*)d_in[0];
  const float* tokens = (const float*)d_in[1];
  const float* gamma  = (const float*)d_in[2];
  const float* w_qkv  = (const float*)d_in[3];
  const float* w_out  = (const float*)d_in[4];
  const int*   modal  = (const int*)d_in[5];
  float* out = (float*)d_out;

  char* ws = (char*)d_ws;
  size_t o_wqkv = 0;                                           // Wqkv^T bf16 [3072][1024]
  size_t o_wout = o_wqkv + (size_t)DMODEL * 3072 * 2;          // Wout^T bf16 [1024][1024]
  size_t o_xn   = o_wout + (size_t)DMODEL * DMODEL * 2;
  size_t o_pos  = o_xn   + (size_t)BATCH * NSEQ * DMODEL * 2;
  size_t o_qkv  = o_pos  + (size_t)BATCH * NSEQ * 4;
  size_t o_q    = o_qkv  + (size_t)BATCH * NSEQ * 3072 * 2;
  size_t o_k    = o_q    + (size_t)BATCH * HEADS * NSEQ * DHEAD * 2;
  size_t o_v    = o_k    + (size_t)BATCH * HEADS * NSEQ * DHEAD * 2;
  size_t o_att  = o_v    + (size_t)BATCH * HEADS * NSEQ * DHEAD * 2;

  bf16* wqkvT = (bf16*)(ws + o_wqkv);
  bf16* woutT = (bf16*)(ws + o_wout);
  bf16* xnb   = (bf16*)(ws + o_xn);
  int*  posb  = (int*)(ws + o_pos);
  bf16* qkvb  = (bf16*)(ws + o_qkv);
  bf16* qb    = (bf16*)(ws + o_q);
  bf16* kb    = (bf16*)(ws + o_k);
  bf16* vtb   = (bf16*)(ws + o_v);
  bf16* attb  = (bf16*)(ws + o_att);

  int nw1 = DMODEL * 3072;
  int nw2 = DMODEL * DMODEL;
  convert_bf16_T<<<(nw1 + 255) / 256, 256, 0, stream>>>(w_qkv, wqkvT, DMODEL, 3072);
  convert_bf16_T<<<(nw2 + 255) / 256, 256, 0, stream>>>(w_out, woutT, DMODEL, DMODEL);

  prep_kernel<<<BATCH * NSEQ, 256, 0, stream>>>(x, tokens, gamma, modal, xnb, posb);

  gemm_bf16_kernel<true><<<dim3(3072 / 128, (BATCH * NSEQ) / 128), 256, 0, stream>>>(
      xnb, wqkvT, qkvb, BATCH * NSEQ, 3072, DMODEL);

  int nrope = BATCH * NSEQ * HEADS * 32;
  rope_split<<<(nrope + 255) / 256, 256, 0, stream>>>(qkvb, posb, qb, kb, vtb);

  attn_kernel<<<dim3(NSEQ / 64, BATCH * HEADS), 128, 0, stream>>>(qb, kb, vtb, modal, attb);

  gemm_bf16_kernel<false><<<dim3(DMODEL / 128, (BATCH * NSEQ) / 128), 256, 0, stream>>>(
      attb, woutT, out, BATCH * NSEQ, DMODEL, DMODEL);
}
